// PtrNet_59476707115125
// MI455X (gfx1250) — matrix operand order repaired and checked by reading, not compiled, not run on hardware
//
#include <hip/hip_runtime.h>
#include <hip/hip_bf16.h>
#include <math.h>

// ---------------------------------------------------------------------------
// Pointer network on MI455X (gfx1250, wave32, WMMA).
// GEMMs use v_wmma_f32_16x16x32_f16 (f16 in, f32 accumulate), software
// pipelined so fragment loads for step k+1 overlap the WMMAs of step k.
// ---------------------------------------------------------------------------

typedef _Float16 half_t;
typedef _Float16 v8h  __attribute__((ext_vector_type(8)));
typedef _Float16 v16h __attribute__((ext_vector_type(16)));
typedef float    v8f  __attribute__((ext_vector_type(8)));

#define PN_B 512
#define PN_S 128
#define PN_D 2
#define PN_H 512

// ---- WMMA fragment loaders (layouts per CDNA5 ISA 7.12.2, 16-bit, wave32) --

// A (16x32, f16): lane<16 -> row m=lane, elems[0..7]=K(k0..k0+7), [8..15]=K(k0+16..k0+23)
//                 lane>=16 -> row m=lane-16, K shifted by +8 / +24.
__device__ __forceinline__ v16h load_a_frag(const half_t* __restrict__ A, int lda,
                                            int m0, int k0, int lane) {
  int m  = lane & 15;
  int kh = (lane >> 4) & 1;
  const half_t* row = A + (size_t)(m0 + m) * (size_t)lda;
  union { v16h v; v8h h[2]; } u;
  u.h[0] = *(const v8h*)(row + k0 + 8 * kh);
  u.h[1] = *(const v8h*)(row + k0 + 16 + 8 * kh);
  return u.v;
}

// B (32x16, f16), with B = W^T so column n of B == row n of W (contiguous in k):
// lane<16 -> col n=lane, elems[0..7]=K(k0..k0+7), [8..15]=K(k0+16..k0+23); lane>=16 -> K shifted by +8 / +24.
__device__ __forceinline__ v16h load_b_frag(const half_t* __restrict__ W, int ldw,
                                            int n0, int k0, int lane) {
  int n  = lane & 15;
  int kh = (lane >> 4) & 1;
  const half_t* row = W + (size_t)(n0 + n) * (size_t)ldw;
  union { v16h v; v8h h[2]; } u;
  u.h[0] = *(const v8h*)(row + k0 + 8 * kh);
  u.h[1] = *(const v8h*)(row + k0 + 16 + 8 * kh);
  return u.v;
}

__device__ __forceinline__ v8f wmma_f16(v16h a, v16h b, v8f c) {
  return __builtin_amdgcn_wmma_f32_16x16x32_f16(false, a, false, b, (short)0, c,
                                                false, false);
}

// ---------------------------------------------------------------------------
// Generic f16 GEMM:  C[M,N] = A[M,K] @ W[N,K]^T + bias[N], C stored f16.
// Block = 128 threads (4 waves). Block tile = 16(M) x 256(N); each wave owns a
// 16x64 strip (4 WMMA n-tiles, A-fragment reused across tiles). Pipelined.
// Requires: M%16==0, N%256==0, K%32==0, K>=64.
// ---------------------------------------------------------------------------
__global__ __launch_bounds__(128) void gemm_f16_kernel(
    const half_t* __restrict__ A, int lda,
    const half_t* __restrict__ W, int ldw,
    const float* __restrict__ bias,
    half_t* __restrict__ C, int ldc, int K) {
  const int m0   = blockIdx.x * 16;
  const int wave = threadIdx.x >> 5;
  const int lane = threadIdx.x & 31;
  const int nw   = blockIdx.y * 256 + wave * 64;

  v8f acc0 = {}, acc1 = {}, acc2 = {}, acc3 = {};

  // prologue: stage k=0 fragments
  v16h a_c  = load_a_frag(A, lda, m0, 0, lane);
  v16h b_c0 = load_b_frag(W, ldw, nw + 0,  0, lane);
  v16h b_c1 = load_b_frag(W, ldw, nw + 16, 0, lane);
  v16h b_c2 = load_b_frag(W, ldw, nw + 32, 0, lane);
  v16h b_c3 = load_b_frag(W, ldw, nw + 48, 0, lane);

  for (int k0 = 32; k0 < K; k0 += 32) {
    if (k0 + 32 < K)  // emits global_prefetch_b8 (CDNA5)
      __builtin_prefetch(A + (size_t)(m0 + (lane & 15)) * (size_t)lda + k0 + 32, 0, 1);
    // issue next-step loads into fresh registers (overlap with WMMAs below)
    v16h a_n  = load_a_frag(A, lda, m0, k0, lane);
    v16h b_n0 = load_b_frag(W, ldw, nw + 0,  k0, lane);
    v16h b_n1 = load_b_frag(W, ldw, nw + 16, k0, lane);
    v16h b_n2 = load_b_frag(W, ldw, nw + 32, k0, lane);
    v16h b_n3 = load_b_frag(W, ldw, nw + 48, k0, lane);
    acc0 = wmma_f16(a_c, b_c0, acc0);
    acc1 = wmma_f16(a_c, b_c1, acc1);
    acc2 = wmma_f16(a_c, b_c2, acc2);
    acc3 = wmma_f16(a_c, b_c3, acc3);
    a_c = a_n; b_c0 = b_n0; b_c1 = b_n1; b_c2 = b_n2; b_c3 = b_n3;
  }
  acc0 = wmma_f16(a_c, b_c0, acc0);
  acc1 = wmma_f16(a_c, b_c1, acc1);
  acc2 = wmma_f16(a_c, b_c2, acc2);
  acc3 = wmma_f16(a_c, b_c3, acc3);

  const int mh = 8 * (lane >> 4);
  const v8f* accs[4] = {&acc0, &acc1, &acc2, &acc3};
#pragma unroll
  for (int nt = 0; nt < 4; ++nt) {
    const int col = nw + nt * 16 + (lane & 15);
    const float bcol = bias ? bias[col] : 0.f;
    const v8f av = *accs[nt];
#pragma unroll
    for (int r = 0; r < 8; ++r) {
      const int m = r + mh;
      C[(size_t)(m0 + m) * (size_t)ldc + col] = (half_t)(av[r] + bcol);
    }
  }
}

// ---------------------------------------------------------------------------
// Fused LSTM step: z = hprev @ Whh^T + x @ Wih^T + bias; cell nonlinearity.
// Block tile: 16 batch rows x 64 hidden cols; wave w computes gate w's strip
// (cols w*H + j0 .. +63) via pipelined WMMA, gates exchanged via LDS, cell.
// ---------------------------------------------------------------------------
__global__ __launch_bounds__(128) void lstm_step_kernel(
    const half_t* __restrict__ hprev, int ldh_in,
    const float* __restrict__ xin, int ldx,          // [B, 2] rows, stride ldx
    const half_t* __restrict__ Whh,                  // [4H, H] f16
    const float* __restrict__ Wih,                   // [4H, 2] f32
    const float* __restrict__ bias,                  // [4H] (bih + bhh)
    float* __restrict__ cbuf,                        // [B, H] f32, in/out
    half_t* __restrict__ hout, int ldh_out) {
  const int H = PN_H;
  __shared__ float zsh[4][16][65];  // +1 pad vs bank conflicts

  const int b0   = blockIdx.x * 16;
  const int j0   = blockIdx.y * 64;
  const int wave = threadIdx.x >> 5;
  const int lane = threadIdx.x & 31;
  const int gb   = wave * H + j0;   // gate column base for this wave

  v8f acc0 = {}, acc1 = {}, acc2 = {}, acc3 = {};

  v16h a_c  = load_a_frag(hprev, ldh_in, b0, 0, lane);
  v16h b_c0 = load_b_frag(Whh, H, gb + 0,  0, lane);
  v16h b_c1 = load_b_frag(Whh, H, gb + 16, 0, lane);
  v16h b_c2 = load_b_frag(Whh, H, gb + 32, 0, lane);
  v16h b_c3 = load_b_frag(Whh, H, gb + 48, 0, lane);

  for (int k0 = 32; k0 < H; k0 += 32) {
    if (k0 + 32 < H)
      __builtin_prefetch(hprev + (size_t)(b0 + (lane & 15)) * (size_t)ldh_in + k0 + 32, 0, 1);
    v16h a_n  = load_a_frag(hprev, ldh_in, b0, k0, lane);
    v16h b_n0 = load_b_frag(Whh, H, gb + 0,  k0, lane);
    v16h b_n1 = load_b_frag(Whh, H, gb + 16, k0, lane);
    v16h b_n2 = load_b_frag(Whh, H, gb + 32, k0, lane);
    v16h b_n3 = load_b_frag(Whh, H, gb + 48, k0, lane);
    acc0 = wmma_f16(a_c, b_c0, acc0);
    acc1 = wmma_f16(a_c, b_c1, acc1);
    acc2 = wmma_f16(a_c, b_c2, acc2);
    acc3 = wmma_f16(a_c, b_c3, acc3);
    a_c = a_n; b_c0 = b_n0; b_c1 = b_n1; b_c2 = b_n2; b_c3 = b_n3;
  }
  acc0 = wmma_f16(a_c, b_c0, acc0);
  acc1 = wmma_f16(a_c, b_c1, acc1);
  acc2 = wmma_f16(a_c, b_c2, acc2);
  acc3 = wmma_f16(a_c, b_c3, acc3);

  const int mh = 8 * (lane >> 4);
  const v8f* accs[4] = {&acc0, &acc1, &acc2, &acc3};
#pragma unroll
  for (int nt = 0; nt < 4; ++nt) {
    const int jloc = nt * 16 + (lane & 15);
    const int gcol = gb + jloc;
    const float bg  = bias[gcol];
    const float wi0 = Wih[gcol * 2 + 0];
    const float wi1 = Wih[gcol * 2 + 1];
    const v8f av = *accs[nt];
#pragma unroll
    for (int r = 0; r < 8; ++r) {
      const int m = r + mh;
      const float x0 = xin[(size_t)(b0 + m) * (size_t)ldx + 0];
      const float x1 = xin[(size_t)(b0 + m) * (size_t)ldx + 1];
      zsh[wave][m][jloc] = av[r] + bg + x0 * wi0 + x1 * wi1;
    }
  }
  __syncthreads();

  for (int p = threadIdx.x; p < 16 * 64; p += 128) {
    const int m = p >> 6, j = p & 63;
    const float zi = zsh[0][m][j];
    const float zf = zsh[1][m][j];
    const float zg = zsh[2][m][j];
    const float zo = zsh[3][m][j];
    const size_t ci = (size_t)(b0 + m) * H + j0 + j;
    const float si = 1.f / (1.f + __expf(-zi));
    const float sf = 1.f / (1.f + __expf(-zf));
    const float so = 1.f / (1.f + __expf(-zo));
    const float cn = sf * cbuf[ci] + si * tanhf(zg);
    cbuf[ci] = cn;
    hout[(size_t)(b0 + m) * (size_t)ldh_out + j0 + j] = (half_t)(so * tanhf(cn));
  }
}

// ---------------------------------------------------------------------------
// Fused attention + softmax + argmax + gather + context, one block (512 thr,
// 16 waves) per batch row b:
//   u[s]  = v_b + sum_h v[h] * tanh(W1e[s,b,h] + q[b,h])   (waves, shfl reduce)
//   a     = softmax(u); out[b,t,:] = a; cur[b] = x[b, argmax u];
//   ctx[b,h] = sum_s a[s] * enc[s,b,h]  -> low half of concat buffer.
// ---------------------------------------------------------------------------
__global__ __launch_bounds__(512) void attn_softmax_ctx_kernel(
    const half_t* __restrict__ W1e,    // [S*B, H] f16 (rows = s*B + b)
    const half_t* __restrict__ q,      // [B, H]   f16
    const float* __restrict__ vw,      // [H]
    const float* __restrict__ vbp,     // [1]
    const float* __restrict__ x,       // [B, S, 2]
    const half_t* __restrict__ ench,   // [S, B, H] f16
    float* __restrict__ outbase, int t,// d_out, write (b, t, :)
    float* __restrict__ cur,           // [B, 2]
    half_t* __restrict__ cat) {        // [B, 2H], write cols [0,H)
  const int S = PN_S, H = PN_H, B = PN_B;
  __shared__ float u_sh[PN_S];
  __shared__ float a_sh[PN_S];
  __shared__ float red[PN_S];
  __shared__ int   redi[PN_S];
  const int b = blockIdx.x, tid = threadIdx.x;
  const int wave = tid >> 5, lane = tid & 31;

  // ---- phase 1: attention scores (16 waves x 8 rows each) ----
  const half_t* qrow = q + (size_t)b * H;
  const float vb = vbp[0];
#pragma unroll
  for (int i = 0; i < 8; ++i) {
    const int s = wave * 8 + i;
    const half_t* w1row = W1e + ((size_t)s * B + b) * H;
    float acc = 0.f;
#pragma unroll 4
    for (int kk = 0; kk < 16; ++kk) {
      const int h = lane + kk * 32;
      acc += vw[h] * tanhf((float)w1row[h] + (float)qrow[h]);
    }
#pragma unroll
    for (int off = 16; off >= 1; off >>= 1) acc += __shfl_xor(acc, off, 32);
    if (lane == 0) u_sh[s] = acc + vb;
  }
  __syncthreads();

  // ---- phase 2: max + argmax (first-index tie-break) ----
  const float val = (tid < S) ? u_sh[tid] : -3.4e38f;
  if (tid < S) { red[tid] = val; redi[tid] = tid; }
  __syncthreads();
  for (int off = 64; off >= 1; off >>= 1) {
    if (tid < off) {
      const float o = red[tid + off];
      const int  oi = redi[tid + off];
      if (o > red[tid] || (o == red[tid] && oi < redi[tid])) {
        red[tid] = o; redi[tid] = oi;
      }
    }
    __syncthreads();
  }
  const float mx = red[0];
  const int chosen = redi[0];
  __syncthreads();

  // ---- phase 3: softmax ----
  const float e = (tid < S) ? __expf(val - mx) : 0.f;
  if (tid < S) red[tid] = e;
  __syncthreads();
  for (int off = 64; off >= 1; off >>= 1) {
    if (tid < off) red[tid] += red[tid + off];
    __syncthreads();
  }
  const float inv = 1.f / red[0];
  if (tid < S) {
    const float a = e * inv;
    a_sh[tid] = a;
    outbase[((size_t)b * S + t) * S + tid] = a;
  }
  if (tid < 2) cur[(size_t)b * 2 + tid] = x[((size_t)b * S + chosen) * 2 + tid];
  __syncthreads();

  // ---- phase 4: context einsum ----
  float acc = 0.f;
  for (int s = 0; s < S; ++s)
    acc += a_sh[s] * (float)ench[((size_t)s * B + b) * H + tid];
  cat[(size_t)b * (2 * H) + tid] = (half_t)acc;
}

// ---------------------------------------------------------------------------
// Small helpers
// ---------------------------------------------------------------------------
__global__ void f32_to_f16_kernel(const float* __restrict__ s,
                                  half_t* __restrict__ d, int n) {
  for (int i = blockIdx.x * blockDim.x + threadIdx.x; i < n;
       i += gridDim.x * blockDim.x)
    d[i] = (half_t)s[i];
}

__global__ void addvec_kernel(const float* __restrict__ a,
                              const float* __restrict__ b,
                              float* __restrict__ o, int n) {
  for (int i = blockIdx.x * blockDim.x + threadIdx.x; i < n;
       i += gridDim.x * blockDim.x)
    o[i] = a[i] + b[i];
}

__global__ void init_kernel(half_t* __restrict__ h0, int n0,
                            float* __restrict__ c, int nc,
                            float* __restrict__ cur, int ncur) {
  const int stride = gridDim.x * blockDim.x;
  for (int i = blockIdx.x * blockDim.x + threadIdx.x; i < n0; i += stride)
    h0[i] = (half_t)0.f;
  for (int i = blockIdx.x * blockDim.x + threadIdx.x; i < nc; i += stride)
    c[i] = 0.f;
  for (int i = blockIdx.x * blockDim.x + threadIdx.x; i < ncur; i += stride)
    cur[i] = -1.f;
}

// ---------------------------------------------------------------------------
// Host orchestration
// ---------------------------------------------------------------------------
extern "C" void kernel_launch(void* const* d_in, const int* in_sizes, int n_in,
                              void* d_out, int out_size, void* d_ws, size_t ws_size,
                              hipStream_t stream) {
  (void)in_sizes; (void)n_in; (void)out_size; (void)ws_size;
  constexpr int B = PN_B, S = PN_S, H = PN_H;

  const float* x        = (const float*)d_in[0];   // [B,S,2]
  const float* enc_Wih  = (const float*)d_in[1];   // [4H,2]
  const float* enc_Whh  = (const float*)d_in[2];   // [4H,H]
  const float* enc_bih  = (const float*)d_in[3];
  const float* enc_bhh  = (const float*)d_in[4];
  const float* dec_Wih  = (const float*)d_in[5];
  const float* dec_Whh  = (const float*)d_in[6];
  const float* dec_bih  = (const float*)d_in[7];
  const float* dec_bhh  = (const float*)d_in[8];
  const float* W1_w     = (const float*)d_in[9];   // [H,H]
  const float* W1_b     = (const float*)d_in[10];
  const float* W2_w     = (const float*)d_in[11];
  const float* W2_b     = (const float*)d_in[12];
  const float* v_w      = (const float*)d_in[13];  // [1,H]
  const float* v_b      = (const float*)d_in[14];  // [1]
  const float* comb_w   = (const float*)d_in[15];  // [H,2H]
  const float* comb_b   = (const float*)d_in[16];
  float* out = (float*)d_out;                      // [B,S,S]

  // -------- workspace carve-out (L2 is 192MB; whole working set ~146MB) ----
  char* base = (char*)d_ws;
  size_t off = 0;
  auto carve = [&](size_t bytes) -> char* {
    off = (off + 255) & ~(size_t)255;
    char* p = base + off;
    off += bytes;
    return p;
  };
  half_t* enc_h   = (half_t*)carve((size_t)S * B * H * 2);   // [S,B,H] encoder states
  half_t* W1e     = (half_t*)carve((size_t)S * B * H * 2);   // [S,B,H] W1 @ enc + b1
  half_t* encWhhH = (half_t*)carve((size_t)4 * H * H * 2);
  half_t* decWhhH = (half_t*)carve((size_t)4 * H * H * 2);
  half_t* W1h     = (half_t*)carve((size_t)H * H * 2);
  half_t* W2h     = (half_t*)carve((size_t)H * H * 2);
  half_t* combH   = (half_t*)carve((size_t)H * 2 * H * 2);
  float*  encbias = (float*)carve((size_t)4 * H * 4);
  float*  decbias = (float*)carve((size_t)4 * H * 4);
  half_t* h0      = (half_t*)carve((size_t)B * H * 2);       // zeros
  float*  cbuf    = (float*)carve((size_t)B * H * 4);
  half_t* catbuf  = (half_t*)carve((size_t)B * 2 * H * 2);   // [ctx | h_dec]
  half_t* qbuf    = (half_t*)carve((size_t)B * H * 2);
  half_t* hcomb   = (half_t*)carve((size_t)B * H * 2);       // decoder carry h
  float*  curbuf  = (float*)carve((size_t)B * 2 * 4);

  // -------- prep: weight conversion, bias fusion, state init --------------
  auto cvt = [&](const float* s, half_t* d, int n) {
    f32_to_f16_kernel<<<(n + 255) / 256, 256, 0, stream>>>(s, d, n);
  };
  cvt(enc_Whh, encWhhH, 4 * H * H);
  cvt(dec_Whh, decWhhH, 4 * H * H);
  cvt(W1_w, W1h, H * H);
  cvt(W2_w, W2h, H * H);
  cvt(comb_w, combH, H * 2 * H);
  addvec_kernel<<<8, 256, 0, stream>>>(enc_bih, enc_bhh, encbias, 4 * H);
  addvec_kernel<<<8, 256, 0, stream>>>(dec_bih, dec_bhh, decbias, 4 * H);
  init_kernel<<<1024, 256, 0, stream>>>(h0, B * H, cbuf, B * H, curbuf, B * 2);

  const dim3 lstmGrid(B / 16, H / 64);   // 32 x 8
  const dim3 gemmHGrid(B / 16, H / 256); // 32 x 2

  // -------- encoder: 128 sequential fused LSTM steps ----------------------
  for (int t = 0; t < S; ++t) {
    const half_t* hprev = (t == 0) ? h0 : enc_h + (size_t)(t - 1) * B * H;
    lstm_step_kernel<<<lstmGrid, 128, 0, stream>>>(
        hprev, H, x + (size_t)t * PN_D, S * PN_D, encWhhH, enc_Wih, encbias,
        cbuf, enc_h + (size_t)t * B * H, H);
  }

  // -------- W1e = enc_h @ W1^T + W1_b  (65536 x 512 x 512) -----------------
  gemm_f16_kernel<<<dim3(S * B / 16, H / 256), 128, 0, stream>>>(
      enc_h, H, W1h, H, W1_b, W1e, H, H);

  // -------- decoder: 128 sequential pointer-attention steps ----------------
  for (int t = 0; t < S; ++t) {
    const half_t* hin = (t == 0) ? enc_h + (size_t)(S - 1) * B * H : hcomb;
    // LSTMCell; h written into high half of concat buffer (cols H..2H-1)
    lstm_step_kernel<<<lstmGrid, 128, 0, stream>>>(
        hin, H, curbuf, 2, decWhhH, dec_Wih, decbias, cbuf, catbuf + H, 2 * H);
    // q = h @ W2^T + b2
    gemm_f16_kernel<<<gemmHGrid, 128, 0, stream>>>(
        catbuf + H, 2 * H, W2h, H, W2_b, qbuf, H, H);
    // attention scores + softmax + argmax + ctx, fused (one block per b)
    attn_softmax_ctx_kernel<<<B, 512, 0, stream>>>(
        W1e, qbuf, v_w, v_b, x, enc_h, out, t, curbuf, catbuf);
    // h_next = [ctx, h] @ comb^T + comb_b
    gemm_f16_kernel<<<gemmHGrid, 128, 0, stream>>>(
        catbuf, 2 * H, combH, 2 * H, comb_b, hcomb, H, 2 * H);
  }
}
